// PolarLayer_24146306138360
// MI455X (gfx1250) — matrix operand order repaired and checked by reading, not compiled, not run on hardware
//
#include <hip/hip_runtime.h>

#define W_IN    8192
#define W_OUT   8190
#define DIM     64
#define FILTERS 32
#define KSZ     3
#define BATCH   128

#define BLOCK_W 128          // w-positions per block (8 waves x 16)
#define X_ROWS  (BLOCK_W + 2)
#define XS      72           // LDS x row stride in bf16 elems (bank spread, 16B align kept)
#define KSTR    200          // LDS kern row stride per filter in bf16 elems

typedef __attribute__((ext_vector_type(16))) __bf16 v16bf;
typedef __attribute__((ext_vector_type(8)))  __bf16 v8bf;
typedef __attribute__((ext_vector_type(8)))  float  v8f;
typedef __attribute__((ext_vector_type(4)))  unsigned uint32x4;
typedef __attribute__((ext_vector_type(8)))  int      int32x8;
typedef __attribute__((ext_vector_type(4)))  int      int32x4;

union Frag { v16bf v; v8bf h[2]; };

// order-preserving float <-> uint encoding for atomic min/max
__device__ __forceinline__ unsigned encf(float f) {
    unsigned u = __float_as_uint(f);
    return u ^ ((u & 0x80000000u) ? 0xFFFFFFFFu : 0x80000000u);
}
__device__ __forceinline__ float decf(unsigned u) {
    unsigned b = (u & 0x80000000u) ? (u ^ 0x80000000u) : ~u;
    return __uint_as_float(b);
}

__global__ void init_minmax(unsigned* __restrict__ wsMin, unsigned* __restrict__ wsMax) {
    int i = threadIdx.x;
    if (i < BATCH) { wsMin[i] = 0xFFFFFFFFu; wsMax[i] = 0u; }
}

__global__ __launch_bounds__(256) void conv_wmma_bf16(
        const float* __restrict__ x, const float* __restrict__ kern,
        float* __restrict__ out,
        unsigned* __restrict__ wsMin, unsigned* __restrict__ wsMax) {
    __shared__ __align__(16) float  xf[X_ROWS * DIM];       // 33,280 B : TDM destination (f32)
    __shared__ __align__(16) __bf16 sx[X_ROWS * XS];        // 18,720 B : bf16 A-operand tile
    __shared__ __align__(16) __bf16 sk[FILTERS * KSTR];     // 12,800 B : bf16 B-operand (f-major)

    const int b   = blockIdx.y;
    const int w0  = blockIdx.x * BLOCK_W;
    const int tid = threadIdx.x;
    const int wave = tid >> 5;
    const int lane = tid & 31;

    // ---- TDM: DMA x rows [w0 .. w0+129] (f32, 130x64 tile) into LDS ----
    if (wave == 0) {
        unsigned long long ga = (unsigned long long)(uintptr_t)x
                              + ((unsigned long long)((size_t)b * W_IN + w0)) * (DIM * 4ull);
        unsigned lds_off = (unsigned)(uintptr_t)(&xf[0]);   // low 32 bits = LDS byte offset
        // D# group 0: count=1 | lds_addr | global_addr[56:0] | type=2
        uint32x4 g0 = { 1u, lds_off, (unsigned)(ga & 0xFFFFFFFFull),
                        (unsigned)((ga >> 32) & 0x01FFFFFFull) | 0x80000000u };
        // D# group 1: mask=0, data_size=2 (4B), tensor_dim0=64,
        // tensor_dim1 = total rows (OOB rows past tensor end read as zero),
        // tile_dim0=64, tile_dim1=130, tensor_dim0_stride=64
        const unsigned td1 = (unsigned)BATCH * W_IN;        // 1,048,576 rows
        int32x8 g1 = { (int)0x00020000u,                    // data_size=4B
                       (int)((unsigned)DIM << 16),          // tensor_dim0=64 (low 16 in [31:16])
                       (int)((td1 & 0xFFFFu) << 16),        // tensor_dim1[15:0]
                       (int)(((td1 >> 16) & 0xFFFFu) | ((unsigned)DIM << 16)), // dim1 hi | tile_dim0
                       (int)X_ROWS,                         // tile_dim1=130, tile_dim2=0
                       (int)DIM,                            // tensor_dim0_stride=64
                       0, 0 };                              // dim1_stride unused (2-D tile)
        int32x4 gz4 = { 0, 0, 0, 0 };
        int32x8 gz8 = { 0, 0, 0, 0, 0, 0, 0, 0 };
        __builtin_amdgcn_tensor_load_to_lds(g0, g1, gz4, gz4, gz8, 0);
    }

    // ---- overlap: stage weights g_kern[(k*64+d)*32+f] -> sk[f*KSTR + K] (bf16) ----
    for (int i = tid; i < KSZ * DIM * FILTERS / 4; i += 256) {   // 1536 float4s
        int e  = i * 4;
        int K  = e >> 5;
        int f0 = e & 31;
        const float4 v = *(const float4*)(kern + e);
        sk[(f0 + 0) * KSTR + K] = (__bf16)v.x;
        sk[(f0 + 1) * KSTR + K] = (__bf16)v.y;
        sk[(f0 + 2) * KSTR + K] = (__bf16)v.z;
        sk[(f0 + 3) * KSTR + K] = (__bf16)v.w;
    }
    __builtin_amdgcn_s_wait_tensorcnt(0);   // no-op for waves with TENSORcnt==0
    __syncthreads();

    // ---- convert staged f32 tile -> bank-padded bf16 tile ----
    for (int i = tid; i < X_ROWS * (DIM / 4); i += 256) {
        int r  = i >> 4;
        int c4 = (i & 15) * 4;
        const float4 v = *(const float4*)(&xf[r * DIM + c4]);
        __bf16* dst = &sx[r * XS + c4];
        dst[0] = (__bf16)v.x; dst[1] = (__bf16)v.y;
        dst[2] = (__bf16)v.z; dst[3] = (__bf16)v.w;
    }
    __syncthreads();

    const int h = lane >> 4;   // half-wave select
    const int m = lane & 15;   // row (A) / col (B,C,D) index

    v8f acc0 = {}; v8f acc1 = {};

    #pragma unroll
    for (int s = 0; s < 6; ++s) {           // K = 192 = 6 x 32
        const int k  = s >> 1;              // conv tap
        const int d0 = (s & 1) * 32;        // channel base
        Frag A, B0, B1;
        const __bf16* ar = &sx[(wave * 16 + m + k) * XS + d0];
        // 16-bit A layout: lanes0-15 VGPR0-3:K0..7, VGPR4-7:K16..23; lanes16-31: +8
        A.h[0]  = *(const v8bf*)(ar + 8 * h);
        A.h[1]  = *(const v8bf*)(ar + 16 + 8 * h);
        // 16-bit B layout: lanes0-15 VGPR0-3:K0..7, VGPR4-7:K16..23; lanes16-31: +8 (col = m)
        const __bf16* b0 = &sk[m * KSTR + 32 * s + 8 * h];
        const __bf16* b1 = &sk[(16 + m) * KSTR + 32 * s + 8 * h];
        B0.h[0] = *(const v8bf*)(b0);  B0.h[1] = *(const v8bf*)(b0 + 16);
        B1.h[0] = *(const v8bf*)(b1);  B1.h[1] = *(const v8bf*)(b1 + 16);

        acc0 = __builtin_amdgcn_wmma_f32_16x16x32_bf16(false, A.v, false, B0.v,
                                                       (short)0, acc0, false, false);
        acc1 = __builtin_amdgcn_wmma_f32_16x16x32_bf16(false, A.v, false, B1.v,
                                                       (short)0, acc1, false, false);
    }

    // ---- store + masked per-lane min/max ----
    const int wbase = w0 + wave * 16;
    float mn =  3.4e38f, mx = -3.4e38f;
    #pragma unroll
    for (int r = 0; r < 8; ++r) {
        int w = wbase + r + 8 * h;          // C/D layout: VGPR r, half h -> M = r + 8h
        if (w < W_OUT) {
            size_t o = ((size_t)b * W_OUT + w) * FILTERS + m;
            out[o]      = acc0[r];
            out[o + 16] = acc1[r];
            mn = fminf(mn, fminf(acc0[r], acc1[r]));
            mx = fmaxf(mx, fmaxf(acc0[r], acc1[r]));
        }
    }
    // wave32 reduction
    #pragma unroll
    for (int off = 16; off; off >>= 1) {
        mn = fminf(mn, __shfl_xor(mn, off, 32));
        mx = fmaxf(mx, __shfl_xor(mx, off, 32));
    }
    if (lane == 0) {
        atomicMin(&wsMin[b], encf(mn));
        atomicMax(&wsMax[b], encf(mx));
    }
}

__global__ __launch_bounds__(256) void normalize_k(
        float* __restrict__ out,
        const unsigned* __restrict__ wsMin, const unsigned* __restrict__ wsMax) {
    const int b = blockIdx.y;
    const float mn  = decf(wsMin[b]);
    const float mx  = decf(wsMax[b]);
    const float inv = 1.0f / (mx - mn);
    const int n4 = W_OUT * FILTERS / 4;     // 65520 float4 per batch
    int i = blockIdx.x * 256 + threadIdx.x;
    if (i < n4) {
        float4* p = (float4*)out + (size_t)b * n4 + i;
        float4 v = *p;
        v.x = (v.x - mn) * inv; v.y = (v.y - mn) * inv;
        v.z = (v.z - mn) * inv; v.w = (v.w - mn) * inv;
        *p = v;
    }
}

extern "C" void kernel_launch(void* const* d_in, const int* in_sizes, int n_in,
                              void* d_out, int out_size, void* d_ws, size_t ws_size,
                              hipStream_t stream) {
    const float* x    = (const float*)d_in[0];
    const float* kern = (const float*)d_in[1];
    float* out        = (float*)d_out;
    unsigned* wsMin   = (unsigned*)d_ws;
    unsigned* wsMax   = wsMin + BATCH;

    init_minmax<<<1, 128, 0, stream>>>(wsMin, wsMax);

    dim3 gConv(W_IN / BLOCK_W, BATCH);          // 64 x 128 blocks
    conv_wmma_bf16<<<gConv, 256, 0, stream>>>(x, kern, out, wsMin, wsMax);

    dim3 gNorm((W_OUT * FILTERS / 4 + 255) / 256, BATCH);
    normalize_k<<<gNorm, 256, 0, stream>>>(out, wsMin, wsMax);
}
